// GNN_50087908606721
// MI455X (gfx1250) — matrix operand order repaired and checked by reading, not compiled, not run on hardware
//
#include <hip/hip_runtime.h>
#include <hip/hip_bf16.h>
#include <math.h>

#define N_NODES 50000
#define M_ENS   20
#define C_INF   32
#define HD      128
#define N_EDGE  800000
#define N_LAYER 3

typedef __attribute__((ext_vector_type(16))) __bf16 v16bf;
typedef __attribute__((ext_vector_type(8)))  float  v8f;
typedef int v4i_vs __attribute__((vector_size(16)));   // matches builtin's pointee type

// ---------------- workspace layout (bytes) ----------------
#define H_OFF    0L                       // h / nf buffer: N*128 f32 = 25.6 MB
#define AGG_OFF  25600000L                // phi-agg / edge-agg: N*128 f32
#define FRAG_OFF 51200000L                // bf16 weight fragments
#define PHI1F 0L            // 1 ktile * 8 nt * 512 bf16 = 8 KB
#define PHI2F 8192L         // 32 KB   (phi w1|w2 contiguous: 40960 B)
#define RHO1F 40960L        // 32 KB
#define RHO2F 73728L        // 32 KB   (rho1|rho2|dr contiguous: 106496 B)
#define DRF   106496L       // 5 ktile -> 40 KB
#define W1F   147456L       // 3 x 32 KB (BN-scaled)
#define W2F   245760L       // 3 x 32 KB
#define BNB_OFF  (FRAG_OFF + 344064L)     // folded BN bias 3*128 f32
#define BNS_OFF  (FRAG_OFF + 345600L)     // BN column scale 3*128 f32

// ---------------- gfx1250 async global->LDS copy (ASYNCcnt path) ----------------
#if defined(__AMDGCN__) && __has_builtin(__builtin_amdgcn_global_load_async_to_lds_b128) && \
    __has_builtin(__builtin_amdgcn_s_wait_asynccnt)
#define USE_ASYNC_LDS 1
#endif

__device__ __forceinline__ void copy_g2l(const void* g, void* l, int bytes, int tid, int nthr) {
#ifdef USE_ASYNC_LDS
  for (int i = tid * 16; i < bytes; i += nthr * 16) {
    __builtin_amdgcn_global_load_async_to_lds_b128(
        (__attribute__((address_space(1))) v4i_vs*)((const char*)g + i),
        (__attribute__((address_space(3))) v4i_vs*)((char*)l + i), 0, 0);
  }
  __builtin_amdgcn_s_wait_asynccnt(0);
#else
  const uint4* s = (const uint4*)g;
  uint4* d = (uint4*)l;
  int n = bytes >> 4;
  for (int i = tid; i < n; i += nthr) d[i] = s[i];
#endif
}

// ---------------- WMMA helpers ----------------
__device__ __forceinline__ v8f wmma_bf16(v16bf a, v16bf b, v8f c) {
  return __builtin_amdgcn_wmma_f32_16x16x32_bf16(false, a, false, b, (short)0, c, false, false);
}
__device__ __forceinline__ v8f v8f_zero() { v8f z = {}; return z; }

// A fragment (16x32 bf16): element e holds K = (e&7) + (e>>3)*16 + 8*half
__device__ __forceinline__ v16bf make_a(const float* __restrict__ row, int hh) {
  v16bf a;
#pragma unroll
  for (int e = 0; e < 16; ++e) {
    int k = (e & 7) + ((e >> 3) << 4) + (hh << 3);
    a[e] = (__bf16)row[k];
  }
  return a;
}
// combined A for GINE: (1+eps)*h + agg
__device__ __forceinline__ v16bf make_a2(const float* __restrict__ hrow,
                                         const float* __restrict__ arow,
                                         float c1, int hh) {
  v16bf a;
#pragma unroll
  for (int e = 0; e < 16; ++e) {
    int k = (e & 7) + ((e >> 3) << 4) + (hh << 3);
    a[e] = (__bf16)(c1 * hrow[k] + arow[k]);
  }
  return a;
}
// B fragment: 512 bf16 per fragment, 16 contiguous per lane (32B)
__device__ __forceinline__ v16bf load_b(const __bf16* __restrict__ frags, int fi, int ln) {
  return *(const v16bf*)(frags + (long)fi * 512 + ln * 16);
}

// ---------------- prep kernels ----------------
__global__ void bn_prep_kernel(const float* __restrict__ g, const float* __restrict__ b,
                               const float* __restrict__ m, const float* __restrict__ v,
                               const float* __restrict__ b1,
                               float* __restrict__ scale, float* __restrict__ bias) {
  int i = blockIdx.x * HD + threadIdx.x;
  float s = g[i] * rsqrtf(v[i] + 1e-5f);
  scale[i] = s;
  bias[i]  = s * (b1[i] - m[i]) + b[i];
}

// pack K x 128 f32 weight into fragment-major bf16; element e of lane holds
// K = kt*32 + (e&7) + 16*(e>>3) + 8*(lane>>4), N = nt*16 + (lane&15)
__global__ void pack_weight_kernel(const float* __restrict__ W, int KT,
                                   const float* __restrict__ colScale,
                                   __bf16* __restrict__ dst) {
  int t = blockIdx.x * blockDim.x + threadIdx.x;
  int total = KT * 8 * 512;
  if (t >= total) return;
  int f = t >> 9, w = t & 511;
  int lane = w >> 4, e = w & 15;
  int kt = f >> 3, nt = f & 7;
  int k = kt * 32 + (e & 7) + ((e >> 3) << 4) + ((lane >> 4) << 3);
  int n = nt * 16 + (lane & 15);
  float val = W[k * HD + n];
  if (colScale) val *= colScale[n];
  dst[t] = (__bf16)val;
}

__global__ void zero_kernel(float* __restrict__ p, long n) {
  long i = (long)blockIdx.x * blockDim.x + threadIdx.x;
  if (i < n) p[i] = 0.f;
}

// ---------------- phi: ensemble MLP + sum over members ----------------
// block = 320 threads (10 waves), 8 nodes (160 ensemble rows) per block
// dyn LDS: weights 40960 B + act 160*128*4 = 81920 B  -> 122880 B
__global__ __launch_bounds__(320) void phi_kernel(
    const float* __restrict__ ens, const float* __restrict__ b1,
    const float* __restrict__ b2, const __bf16* __restrict__ wfr,
    float* __restrict__ agg) {
  extern __shared__ char smem[];
  __bf16* wsh = (__bf16*)smem;              // w1 frags @0 (4096 el), w2 @4096
  float*  act = (float*)(smem + 40960);
  int tid = threadIdx.x;
  copy_g2l(wfr, wsh, 40960, tid, 320);
  __syncthreads();
  int wv = tid >> 5, ln = tid & 31, m = ln & 15, hh = ln >> 4;
  int rowbase = wv * 16;
  long g = (long)blockIdx.x * 160 + rowbase + m;  // (node*20 + member), contiguous

  // GEMM1: (16x32) @ (32x128) + b1, relu
  v16bf a = make_a(ens + g * C_INF, hh);
  v8f acc[8];
#pragma unroll
  for (int nt = 0; nt < 8; ++nt) acc[nt] = v8f_zero();
#pragma unroll
  for (int nt = 0; nt < 8; ++nt) acc[nt] = wmma_bf16(a, load_b(wsh, nt, ln), acc[nt]);
#pragma unroll
  for (int nt = 0; nt < 8; ++nt) {
    int col = nt * 16 + m;
    float bv = b1[col];
#pragma unroll
    for (int d = 0; d < 8; ++d)
      act[(rowbase + d + 8 * hh) * HD + col] = fmaxf(acc[nt][d] + bv, 0.f);
  }
  // GEMM2: (16x128) @ (128x128) + b2
  v8f acc2[8];
#pragma unroll
  for (int nt = 0; nt < 8; ++nt) acc2[nt] = v8f_zero();
#pragma unroll
  for (int kt = 0; kt < 4; ++kt) {
    v16bf a2 = make_a(act + (rowbase + m) * HD + kt * 32, hh);
#pragma unroll
    for (int nt = 0; nt < 8; ++nt)
      acc2[nt] = wmma_bf16(a2, load_b(wsh + 4096, kt * 8 + nt, ln), acc2[nt]);
  }
#pragma unroll
  for (int nt = 0; nt < 8; ++nt) {
    int col = nt * 16 + m;
    float bv = b2[col];
#pragma unroll
    for (int d = 0; d < 8; ++d)
      act[(rowbase + d + 8 * hh) * HD + col] = acc2[nt][d] + bv;
  }
  __syncthreads();
  // sum 20 member rows per node (8 nodes * 128 cols)
  for (int c = tid; c < 8 * HD; c += 320) {
    int j = c >> 7, col = c & 127;
    float s = 0.f;
#pragma unroll
    for (int q = 0; q < M_ENS; ++q) s += act[(j * M_ENS + q) * HD + col];
    agg[((long)blockIdx.x * 8 + j) * HD + col] = s;
  }
}

// ---------------- rho MLP + concat + dr projection ----------------
// block = 256 threads (8 waves), 128 nodes per block
// dyn LDS: weights 106496 B + act 8*16*160*4 = 81920 B -> 188416 B
__global__ __launch_bounds__(256) void rho_dr_kernel(
    const float* __restrict__ aggIn, const float* __restrict__ x,
    const float* __restrict__ rb1, const float* __restrict__ rb2,
    const float* __restrict__ drb, const __bf16* __restrict__ wfr,
    float* __restrict__ hOut) {
  extern __shared__ char smem[];
  __bf16* wsh = (__bf16*)smem;   // rho1 @0, rho2 @16384 el, dr @32768 el
  float*  act = (float*)(smem + 106496);
  int tid = threadIdx.x;
  copy_g2l(wfr, wsh, 106496, tid, 256);
  __syncthreads();
  int wv = tid >> 5, ln = tid & 31, m = ln & 15, hh = ln >> 4;
  long nb = (long)blockIdx.x * 128 + wv * 16;
  long node = nb + m;
  long srcn = (node < N_NODES) ? node : (N_NODES - 1);
  float* st = act + wv * 16 * 160;

  // stage1: relu(agg @ rho_w1 + rb1) -> st cols 0..127
  v8f acc[8];
#pragma unroll
  for (int nt = 0; nt < 8; ++nt) acc[nt] = v8f_zero();
#pragma unroll
  for (int kt = 0; kt < 4; ++kt) {
    v16bf a = make_a(aggIn + srcn * HD + kt * 32, hh);
#pragma unroll
    for (int nt = 0; nt < 8; ++nt)
      acc[nt] = wmma_bf16(a, load_b(wsh, kt * 8 + nt, ln), acc[nt]);
  }
#pragma unroll
  for (int nt = 0; nt < 8; ++nt) {
    int col = nt * 16 + m;
    float bv = rb1[col];
#pragma unroll
    for (int d = 0; d < 8; ++d)
      st[(d + 8 * hh) * 160 + col] = fmaxf(acc[nt][d] + bv, 0.f);
  }
  // stage2: emb = z1 @ rho_w2 + rb2 -> st cols 32..159
  v8f acc2[8];
#pragma unroll
  for (int nt = 0; nt < 8; ++nt) acc2[nt] = v8f_zero();
#pragma unroll
  for (int kt = 0; kt < 4; ++kt) {
    v16bf a = make_a(st + m * 160 + kt * 32, hh);
#pragma unroll
    for (int nt = 0; nt < 8; ++nt)
      acc2[nt] = wmma_bf16(a, load_b(wsh + 16384, kt * 8 + nt, ln), acc2[nt]);
  }
#pragma unroll
  for (int nt = 0; nt < 8; ++nt) {
    int col = nt * 16 + m;
    float bv = rb2[col];
#pragma unroll
    for (int d = 0; d < 8; ++d)
      st[(d + 8 * hh) * 160 + 32 + col] = acc2[nt][d] + bv;
  }
  // fill x into st cols 0..31 (after stage2 reads in program order)
  for (int i = ln; i < 16 * 32; i += 32) {
    int rr = i >> 5, cc = i & 31;
    long nd = nb + rr;
    st[rr * 160 + cc] = (nd < N_NODES) ? x[nd * C_INF + cc] : 0.f;
  }
  // stage3: nf = concat(x, emb) @ dr_w + drb   (K = 160 -> 5 ktiles)
  v8f acc3[8];
#pragma unroll
  for (int nt = 0; nt < 8; ++nt) acc3[nt] = v8f_zero();
#pragma unroll
  for (int kt = 0; kt < 5; ++kt) {
    v16bf a = make_a(st + m * 160 + kt * 32, hh);
#pragma unroll
    for (int nt = 0; nt < 8; ++nt)
      acc3[nt] = wmma_bf16(a, load_b(wsh + 32768, kt * 8 + nt, ln), acc3[nt]);
  }
#pragma unroll
  for (int nt = 0; nt < 8; ++nt) {
    int col = nt * 16 + m;
    float bv = drb[col];
#pragma unroll
    for (int d = 0; d < 8; ++d) {
      long nd = nb + d + 8 * hh;
      if (nd < N_NODES) hOut[nd * HD + col] = acc3[nt][d] + bv;
    }
  }
}

// ---------------- edge messages + scatter-add ----------------
// one wave per edge; lane covers 4 features
__global__ __launch_bounds__(256) void edge_kernel(
    const float* __restrict__ h, const int* __restrict__ ei,
    const float* __restrict__ ea, const float* __restrict__ ew,
    const float* __restrict__ eb, float* __restrict__ agg) {
  int wid = (blockIdx.x * blockDim.x + threadIdx.x) >> 5;
  int ln = threadIdx.x & 31;
  if (wid >= N_EDGE) return;
  int s = ei[wid];
  int d = ei[N_EDGE + wid];
  float a = ea[wid];
  float4 hv = ((const float4*)(h + (long)s * HD))[ln];
  float4 wv = ((const float4*)ew)[ln];
  float4 bv = ((const float4*)eb)[ln];
  float4 mv;
  mv.x = fmaxf(hv.x + a * wv.x + bv.x, 0.f);
  mv.y = fmaxf(hv.y + a * wv.y + bv.y, 0.f);
  mv.z = fmaxf(hv.z + a * wv.z + bv.z, 0.f);
  mv.w = fmaxf(hv.w + a * wv.w + bv.w, 0.f);
  float* ag = agg + (long)d * HD + ln * 4;
  unsafeAtomicAdd(ag + 0, mv.x);
  unsafeAtomicAdd(ag + 1, mv.y);
  unsafeAtomicAdd(ag + 2, mv.z);
  unsafeAtomicAdd(ag + 3, mv.w);
}

// ---------------- GINE MLP: z=(1+eps)h+agg; BN-folded w1; relu; w2; residual ----------------
// block = 256 threads (8 waves), 128 nodes; dyn LDS: 65536 + 65536 = 131072 B
__global__ __launch_bounds__(256) void gine_mlp_kernel(
    float* __restrict__ h, const float* __restrict__ agg,
    const float* __restrict__ epsArr, int layer,
    const float* __restrict__ bnbias, const float* __restrict__ b2,
    const __bf16* __restrict__ w1f, const __bf16* __restrict__ w2f,
    int firstLayer) {
  extern __shared__ char smem[];
  __bf16* wsh = (__bf16*)smem;   // w1 @0, w2 @16384 el
  float*  act = (float*)(smem + 65536);
  int tid = threadIdx.x;
  copy_g2l(w1f, wsh, 32768, tid, 256);
  copy_g2l(w2f, wsh + 16384, 32768, tid, 256);
  __syncthreads();
  int wv = tid >> 5, ln = tid & 31, m = ln & 15, hh = ln >> 4;
  long nb = (long)blockIdx.x * 128 + wv * 16;
  long node = nb + m;
  long srcn = (node < N_NODES) ? node : (N_NODES - 1);
  float c1 = 1.f + epsArr[layer];
  float* st = act + wv * 16 * HD;

  // stage1: u = relu( ((1+eps)h+agg) @ (w1*s) + bias' )
  v8f acc[8];
#pragma unroll
  for (int nt = 0; nt < 8; ++nt) acc[nt] = v8f_zero();
#pragma unroll
  for (int kt = 0; kt < 4; ++kt) {
    v16bf a = make_a2(h + srcn * HD + kt * 32, agg + srcn * HD + kt * 32, c1, hh);
#pragma unroll
    for (int nt = 0; nt < 8; ++nt)
      acc[nt] = wmma_bf16(a, load_b(wsh, kt * 8 + nt, ln), acc[nt]);
  }
#pragma unroll
  for (int nt = 0; nt < 8; ++nt) {
    int col = nt * 16 + m;
    float bv = bnbias[col];
#pragma unroll
    for (int d = 0; d < 8; ++d)
      st[(d + 8 * hh) * HD + col] = fmaxf(acc[nt][d] + bv, 0.f);
  }
  // stage2: c = u @ w2 + b2; h = (first? 0 : h) + relu(c)
  v8f acc2[8];
#pragma unroll
  for (int nt = 0; nt < 8; ++nt) acc2[nt] = v8f_zero();
#pragma unroll
  for (int kt = 0; kt < 4; ++kt) {
    v16bf a = make_a(st + m * HD + kt * 32, hh);
#pragma unroll
    for (int nt = 0; nt < 8; ++nt)
      acc2[nt] = wmma_bf16(a, load_b(wsh + 16384, kt * 8 + nt, ln), acc2[nt]);
  }
#pragma unroll
  for (int nt = 0; nt < 8; ++nt) {
    int col = nt * 16 + m;
    float bv = b2[col];
#pragma unroll
    for (int d = 0; d < 8; ++d) {
      long nd = nb + d + 8 * hh;
      if (nd < N_NODES) {
        float rc = fmaxf(acc2[nt][d] + bv, 0.f);
        float hv = firstLayer ? 0.f : h[nd * HD + col];
        h[nd * HD + col] = hv + rc;  // in-place: each row owned by this wave
      }
    }
  }
}

// ---------------- final head: h @ aggr_w + aggr_b; softplus on col1 ----------------
__global__ __launch_bounds__(256) void final_kernel(
    const float* __restrict__ h, const float* __restrict__ aw,
    const float* __restrict__ ab, float* __restrict__ out) {
  int wv = threadIdx.x >> 5, ln = threadIdx.x & 31;
  long base = ((long)blockIdx.x * 8 + wv) * 32;
  float wmu[4], wsg[4];
#pragma unroll
  for (int j = 0; j < 4; ++j) {
    int k = ln * 4 + j;
    wmu[j] = aw[k * 2];
    wsg[j] = aw[k * 2 + 1];
  }
  for (int j = 0; j < 32; ++j) {
    long node = base + j;
    if (node >= N_NODES) break;
    float4 hv = ((const float4*)(h + node * HD))[ln];
    float p0 = hv.x * wmu[0] + hv.y * wmu[1] + hv.z * wmu[2] + hv.w * wmu[3];
    float p1 = hv.x * wsg[0] + hv.y * wsg[1] + hv.z * wsg[2] + hv.w * wsg[3];
#pragma unroll
    for (int o = 16; o > 0; o >>= 1) {
      p0 += __shfl_xor(p0, o, 32);
      p1 += __shfl_xor(p1, o, 32);
    }
    if (ln == 0) {
      float mu = p0 + ab[0];
      float sg = p1 + ab[1];
      float sp = (sg > 20.f) ? sg : log1pf(expf(sg));
      out[node * 2]     = mu;
      out[node * 2 + 1] = sp;
    }
  }
}

// ---------------- launch ----------------
extern "C" void kernel_launch(void* const* d_in, const int* in_sizes, int n_in,
                              void* d_out, int out_size, void* d_ws, size_t ws_size,
                              hipStream_t stream) {
  const float* ens  = (const float*)d_in[0];
  const float* x    = (const float*)d_in[1];
  const int*   ei   = (const int*)d_in[2];
  const float* ea   = (const float*)d_in[3];
  const float* pw1  = (const float*)d_in[4];
  const float* pb1  = (const float*)d_in[5];
  const float* pw2  = (const float*)d_in[6];
  const float* pb2  = (const float*)d_in[7];
  const float* rw1  = (const float*)d_in[8];
  const float* rb1  = (const float*)d_in[9];
  const float* rw2  = (const float*)d_in[10];
  const float* rb2  = (const float*)d_in[11];
  const float* drw  = (const float*)d_in[12];
  const float* drb  = (const float*)d_in[13];
  const float* ceps = (const float*)d_in[14];
  const float* ew   = (const float*)d_in[15];
  const float* eb   = (const float*)d_in[16];
  const float* mw1  = (const float*)d_in[17];
  const float* mb1  = (const float*)d_in[18];
  const float* bng  = (const float*)d_in[19];
  const float* bnb  = (const float*)d_in[20];
  const float* bnm  = (const float*)d_in[21];
  const float* bnv  = (const float*)d_in[22];
  const float* mw2  = (const float*)d_in[23];
  const float* mb2  = (const float*)d_in[24];
  const float* aw   = (const float*)d_in[25];
  const float* ab   = (const float*)d_in[26];
  (void)in_sizes; (void)n_in; (void)out_size; (void)ws_size;

  char* ws = (char*)d_ws;
  float*  bufH    = (float*)(ws + H_OFF);
  float*  bufA    = (float*)(ws + AGG_OFF);
  float*  bnbias  = (float*)(ws + BNB_OFF);
  float*  bnscale = (float*)(ws + BNS_OFF);
  __bf16* fr      = (__bf16*)(ws + FRAG_OFF);

  // fold BN into scale/bias
  bn_prep_kernel<<<N_LAYER, HD, 0, stream>>>(bng, bnb, bnm, bnv, mb1, bnscale, bnbias);

  // pack weights to bf16 fragment layout
  auto nb = [](int KT) { return (KT * 8 * 512 + 255) / 256; };
  pack_weight_kernel<<<nb(1), 256, 0, stream>>>(pw1, 1, nullptr, (__bf16*)((char*)fr + PHI1F));
  pack_weight_kernel<<<nb(4), 256, 0, stream>>>(pw2, 4, nullptr, (__bf16*)((char*)fr + PHI2F));
  pack_weight_kernel<<<nb(4), 256, 0, stream>>>(rw1, 4, nullptr, (__bf16*)((char*)fr + RHO1F));
  pack_weight_kernel<<<nb(4), 256, 0, stream>>>(rw2, 4, nullptr, (__bf16*)((char*)fr + RHO2F));
  pack_weight_kernel<<<nb(5), 256, 0, stream>>>(drw, 5, nullptr, (__bf16*)((char*)fr + DRF));
  for (int l = 0; l < N_LAYER; ++l) {
    pack_weight_kernel<<<nb(4), 256, 0, stream>>>(mw1 + (long)l * HD * HD, 4, bnscale + l * HD,
                                                  (__bf16*)((char*)fr + W1F + l * 32768L));
    pack_weight_kernel<<<nb(4), 256, 0, stream>>>(mw2 + (long)l * HD * HD, 4, nullptr,
                                                  (__bf16*)((char*)fr + W2F + l * 32768L));
  }

  // phi MLP over ensemble + member-sum: 6250 blocks x 8 nodes, 120 KB dyn LDS
  phi_kernel<<<N_NODES / 8, 320, 122880, stream>>>(
      ens, pb1, pb2, (const __bf16*)((char*)fr + PHI1F), bufA);

  // rho MLP + concat + dr projection -> h: 184 KB dyn LDS
  rho_dr_kernel<<<(N_NODES + 127) / 128, 256, 188416, stream>>>(
      bufA, x, rb1, rb2, drb, (const __bf16*)((char*)fr + RHO1F), bufH);

  // 3 GINE layers
  for (int l = 0; l < N_LAYER; ++l) {
    zero_kernel<<<((long)N_NODES * HD + 255) / 256, 256, 0, stream>>>(bufA, (long)N_NODES * HD);
    edge_kernel<<<(N_EDGE * 32) / 256, 256, 0, stream>>>(bufH, ei, ea, ew + l * HD, eb + l * HD, bufA);
    gine_mlp_kernel<<<(N_NODES + 127) / 128, 256, 131072, stream>>>(
        bufH, bufA, ceps, l, bnbias + l * HD, mb2 + l * HD,
        (const __bf16*)((char*)fr + W1F + l * 32768L),
        (const __bf16*)((char*)fr + W2F + l * 32768L), l == 0 ? 1 : 0);
  }

  // final head
  final_kernel<<<(N_NODES + 255) / 256, 256, 0, stream>>>(bufH, aw, ab, (float*)d_out);
}
